// GATModel_59081570124181
// MI455X (gfx1250) — matrix operand order repaired and checked by reading, not compiled, not run on hardware
//
#include <hip/hip_runtime.h>
#include <hip/hip_bf16.h>

typedef __attribute__((ext_vector_type(16))) _Float16 v16h;
typedef __attribute__((ext_vector_type(8)))  float    v8f;

#define N_NODES 4096
#define F_IN    512
#define CHID    64
#define NHEAD   8
#define CPH     8     // channels per head, layer 1
#define NCLS    7
#define ADJ_W   128   // 4096 bits / 32 per row

__device__ __forceinline__ float lrelu(float v) { return v > 0.f ? v : 0.2f * v; }

// CDNA5 async global->LDS copy (ASYNCcnt-tracked, bypasses VGPRs).
// LDS offset = low 32 bits of the generic pointer.
__device__ __forceinline__ void async_cp16(void* lds_ptr, const void* gptr) {
    unsigned lds_off = (unsigned)(unsigned long long)lds_ptr;
    asm volatile("global_load_async_to_lds_b128 %0, %1, off"
                 :: "v"(lds_off), "v"(gptr)
                 : "memory");
}
__device__ __forceinline__ void async_wait_all() {
    asm volatile("s_wait_asynccnt 0x0" ::: "memory");
}

// ---------------------------------------------------------------- adjacency
__global__ void k_zero(unsigned* __restrict__ p, int n) {
    int t = blockIdx.x * blockDim.x + threadIdx.x;
    if (t < n) p[t] = 0u;
}

__global__ void k_adj(unsigned* __restrict__ adj, const int* __restrict__ e0,
                      const int* __restrict__ e1, int E) {
    int t = blockIdx.x * blockDim.x + threadIdx.x;
    if (t < E) {
        int r = e0[t], c = e1[t];
        atomicOr(&adj[r * ADJ_W + (c >> 5)], 1u << (c & 31));
    }
    if (t < N_NODES) {  // diagonal always unmasked
        atomicOr(&adj[t * ADJ_W + (t >> 5)], 1u << (t & 31));
    }
}

// ---------------------------------------------------------------- W1 -> f16 B-fragment layout
// packed[((kt*4+nt)*32+lane)*16+half]; col = nt*16 + (lane&15); k = kt*32 + (half&7) + 8*(lane>=16) + 16*(half>=8)
__global__ void k_pack_w1(const float* __restrict__ W1, _Float16* __restrict__ W1p) {
    int t = blockIdx.x * blockDim.x + threadIdx.x;          // 32768
    int half = t & 15, lane = (t >> 4) & 31, nt = (t >> 9) & 3, kt = t >> 11;
    int n = nt * 16 + (lane & 15);
    int k = kt * 32 + (half & 7) + ((half >> 3) << 4) + ((lane >> 4) << 3);
    W1p[t] = (_Float16)W1[k * CHID + n];
}

// ---------------------------------------------------------------- GEMM1: feats1 = x@W1 + b1 (f32 out)
__global__ __launch_bounds__(256)
void k_gemm1(const float* __restrict__ x, const _Float16* __restrict__ W1p,
             const float* __restrict__ b1, float* __restrict__ feats1) {
    const int lane = threadIdx.x & 31;
    const int wave = threadIdx.x >> 5;
    const int mrow = (blockIdx.x * 8 + wave) * 16;
    const int sub  = lane >> 4;                 // 0: rows M/K group 0, 1: +8/+8 group
    const int row  = mrow + (lane & 15);
    v8f acc[4] = {};
    const float* xr = x + row * F_IN + sub * 8; // A layout: halves 0-7 -> K=kb..kb+7, 8-15 -> K=kb+16..kb+23
    for (int kt = 0; kt < F_IN / 32; ++kt) {
        const float4 f0 = *(const float4*)(xr + kt * 32 + 0);
        const float4 f1 = *(const float4*)(xr + kt * 32 + 4);
        const float4 f2 = *(const float4*)(xr + kt * 32 + 16);
        const float4 f3 = *(const float4*)(xr + kt * 32 + 20);
        v16h a;
        a[0]=(_Float16)f0.x; a[1]=(_Float16)f0.y; a[2]=(_Float16)f0.z; a[3]=(_Float16)f0.w;
        a[4]=(_Float16)f1.x; a[5]=(_Float16)f1.y; a[6]=(_Float16)f1.z; a[7]=(_Float16)f1.w;
        a[8]=(_Float16)f2.x; a[9]=(_Float16)f2.y; a[10]=(_Float16)f2.z; a[11]=(_Float16)f2.w;
        a[12]=(_Float16)f3.x; a[13]=(_Float16)f3.y; a[14]=(_Float16)f3.z; a[15]=(_Float16)f3.w;
#pragma unroll
        for (int nt = 0; nt < 4; ++nt) {
            const v16h b = *(const v16h*)(W1p + ((kt * 4 + nt) * 32 + lane) * 16);
            acc[nt] = __builtin_amdgcn_wmma_f32_16x16x32_f16(
                false, a, false, b, (short)0, acc[nt], false, false);
        }
    }
#pragma unroll
    for (int nt = 0; nt < 4; ++nt)
#pragma unroll
        for (int r = 0; r < 8; ++r) {
            int ro  = mrow + r + sub * 8;       // D: VGPR r, lanes16-31 -> M+8
            int col = nt * 16 + (lane & 15);
            feats1[ro * CHID + col] = acc[nt][r] + b1[col];
        }
}

// ---------------------------------------------------------------- per-node per-head scores
// ssrcT is transposed: [head][node] so attention blocks can bulk-stage it into LDS.
__global__ void k_scores1(const float* __restrict__ feats1, const float* __restrict__ aL,
                          const float* __restrict__ aR, float* __restrict__ ssrcT,
                          float* __restrict__ stgt) {
    int t = blockIdx.x * blockDim.x + threadIdx.x;          // 32768
    int h = t & 7, i = t >> 3;
    float a = 0.f, b = 0.f;
#pragma unroll
    for (int c = 0; c < CPH; ++c) {
        float f = feats1[i * CHID + h * CPH + c];
        a += f * aL[c * NHEAD + h];
        b += f * aR[c * NHEAD + h];
    }
    ssrcT[h * N_NODES + i] = a;
    stgt[i * NHEAD + h] = b;
}

// feats1 -> f16 B fragments per head (8 valid cols, 8 zero-padded)
__global__ void k_pack_f1(const float* __restrict__ feats1, _Float16* __restrict__ f1p) {
    int t = blockIdx.x * blockDim.x + threadIdx.x;          // 524288
    int half = t & 15, lane = (t >> 4) & 31, h = (t >> 9) & 7, kt = t >> 12;
    int col = lane & 15;
    int j = kt * 32 + (half & 7) + ((half >> 3) << 4) + ((lane >> 4) << 3);
    float v = (col < CPH) ? feats1[j * CHID + h * CPH + col] : 0.f;
    f1p[t] = (_Float16)v;
}

// ---------------------------------------------------------------- layer-1 attention + ELU
// block = 8 waves; all waves share one 16-row tile; wave = head.
// LDS: full transposed source-score table (8 x 4096 f32 = 128 KB) + 16-row adjacency slab (8 KB),
// staged with CDNA5 async global->LDS copies.
__global__ __launch_bounds__(256)
void k_attn1(const unsigned* __restrict__ adj, const float* __restrict__ ssrcT,
             const float* __restrict__ stgt, const _Float16* __restrict__ f1p,
             float* __restrict__ h1) {
    __shared__ __align__(16) float    s_lds[NHEAD * N_NODES];   // 128 KB
    __shared__ __align__(16) unsigned adjs[16 * ADJ_W];         // 8 KB
    const int lane = threadIdx.x & 31;
    const int head = threadIdx.x >> 5;
    const int rbase = blockIdx.x * 16;
    // 8192 + 512 16B chunks, both multiples of 256 -> EXEC full on every issue
    for (int t = threadIdx.x; t < (NHEAD * N_NODES * 4) / 16; t += 256)
        async_cp16((char*)s_lds + t * 16, (const char*)ssrcT + t * 16);
    for (int t = threadIdx.x; t < (16 * ADJ_W * 4) / 16; t += 256)
        async_cp16((char*)adjs + t * 16, (const char*)(adj + rbase * ADJ_W) + t * 16);
    async_wait_all();
    __syncthreads();

    const int rl = lane & 15;
    const float st = stgt[(rbase + rl) * NHEAD + head];
    const float* sh = &s_lds[head * N_NODES];
    // pass 1: row max over unmasked j. Word-wise with zero-word skip (~0.4% density).
    float m = -3.0e38f;
    for (int wi = (lane >> 4); wi < ADJ_W; wi += 2) {
        unsigned w = adjs[rl * ADJ_W + wi];
        if (w == 0u) continue;
        const float* sb = sh + wi * 32;
        for (int b = 0; b < 32; ++b) {
            float s = lrelu(sb[b] + st);
            m = ((w >> b) & 1u) ? fmaxf(m, s) : m;
        }
    }
    m = fmaxf(m, __shfl_xor(m, 16));
    // pass 2: row sum of exp
    float sum = 0.f;
    for (int wi = (lane >> 4); wi < ADJ_W; wi += 2) {
        unsigned w = adjs[rl * ADJ_W + wi];
        if (w == 0u) continue;
        const float* sb = sh + wi * 32;
        for (int b = 0; b < 32; ++b) {
            float s = lrelu(sb[b] + st);
            sum += ((w >> b) & 1u) ? __expf(s - m) : 0.f;
        }
    }
    sum += __shfl_xor(sum, 16);
    const float inv = 1.f / sum;
    // pass 3: WMMA aggregation out = softmax(scores) @ feats (A = probs, built on the fly).
    // A 32-wide j-tile maps to exactly one adjacency word per row (word index == jt).
    v8f acc = {};
    const int koff = (lane >> 4) << 3;
    for (int jt = 0; jt < N_NODES / 32; ++jt) {
        const unsigned w = adjs[rl * ADJ_W + jt];
        if (__ballot(w != 0u) == 0ull) continue;            // wave-uniform skip: EXEC stays full
        const float4* sv = (const float4*)(sh + jt * 32 + koff);
        const float4 s0 = sv[0], s1 = sv[1];                // K = koff .. koff+7
        const float4 s2 = sv[4], s3 = sv[5];                // K = koff+16 .. koff+23
        float sarr[16] = { s0.x, s0.y, s0.z, s0.w, s1.x, s1.y, s1.z, s1.w,
                           s2.x, s2.y, s2.z, s2.w, s3.x, s3.y, s3.z, s3.w };
        v16h a;
#pragma unroll
        for (int hh = 0; hh < 16; ++hh) {
            int K = hh + ((hh >> 3) << 3) + koff;           // A-fragment K mapping
            float s = lrelu(sarr[hh] + st);
            float p = ((w >> K) & 1u) ? __expf(s - m) * inv : 0.f;  // selects only
            a[hh] = (_Float16)p;
        }
        const v16h b = *(const v16h*)(f1p + ((jt * 8 + head) * 32 + lane) * 16);
        acc = __builtin_amdgcn_wmma_f32_16x16x32_f16(
            false, a, false, b, (short)0, acc, false, false);
    }
    const int col = lane & 15;
    if (col < CPH) {
#pragma unroll
        for (int r = 0; r < 8; ++r) {
            int ro = rbase + r + ((lane >> 4) << 3);
            float v = acc[r];
            h1[ro * CHID + head * CPH + col] = v > 0.f ? v : (__expf(v) - 1.f);  // ELU
        }
    }
}

// ---------------------------------------------------------------- layer-2 GEMM (tiny) + scores
__global__ void k_gemm2(const float* __restrict__ h1, const float* __restrict__ W2,
                        const float* __restrict__ b2, const float* __restrict__ aL2,
                        const float* __restrict__ aR2, float* __restrict__ feats2,
                        float* __restrict__ ssrc2, float* __restrict__ stgt2) {
    int i = blockIdx.x * blockDim.x + threadIdx.x;
    if (i >= N_NODES) return;
    float f[NCLS];
#pragma unroll
    for (int c = 0; c < NCLS; ++c) f[c] = b2[c];
    for (int k = 0; k < CHID; ++k) {
        float hv = h1[i * CHID + k];
#pragma unroll
        for (int c = 0; c < NCLS; ++c) f[c] += hv * W2[k * NCLS + c];
    }
    float a = 0.f, r = 0.f;
#pragma unroll
    for (int c = 0; c < NCLS; ++c) {
        feats2[i * NCLS + c] = f[c];
        a += f[c] * aL2[c];
        r += f[c] * aR2[c];
    }
    ssrc2[i] = a;
    stgt2[i] = r;
}

__global__ void k_pack_f2(const float* __restrict__ feats2, _Float16* __restrict__ f2p) {
    int t = blockIdx.x * blockDim.x + threadIdx.x;          // 65536
    int half = t & 15, lane = (t >> 4) & 31, kt = t >> 9;
    int col = lane & 15;
    int j = kt * 32 + (half & 7) + ((half >> 3) << 4) + ((lane >> 4) << 3);
    float v = (col < NCLS) ? feats2[j * NCLS + col] : 0.f;
    f2p[t] = (_Float16)v;
}

// ---------------------------------------------------------------- layer-2 attention (1 head)
__global__ __launch_bounds__(32)
void k_attn2(const unsigned* __restrict__ adj, const float* __restrict__ ssrc,
             const float* __restrict__ stgt, const _Float16* __restrict__ f2p,
             float* __restrict__ out) {
    __shared__ __align__(16) float    s_lds[N_NODES];       // 16 KB
    __shared__ __align__(16) unsigned adjs[16 * ADJ_W];     // 8 KB
    const int lane = threadIdx.x;
    const int rbase = blockIdx.x * 16;
    for (int t = lane; t < (N_NODES * 4) / 16; t += 32)     // 1024 chunks
        async_cp16((char*)s_lds + t * 16, (const char*)ssrc + t * 16);
    for (int t = lane; t < (16 * ADJ_W * 4) / 16; t += 32)  // 512 chunks
        async_cp16((char*)adjs + t * 16, (const char*)(adj + rbase * ADJ_W) + t * 16);
    async_wait_all();
    __syncthreads();

    const int rl = lane & 15;
    const float st = stgt[rbase + rl];
    float m = -3.0e38f;
    for (int wi = (lane >> 4); wi < ADJ_W; wi += 2) {
        unsigned w = adjs[rl * ADJ_W + wi];
        if (w == 0u) continue;
        const float* sb = s_lds + wi * 32;
        for (int b = 0; b < 32; ++b) {
            float s = lrelu(sb[b] + st);
            m = ((w >> b) & 1u) ? fmaxf(m, s) : m;
        }
    }
    m = fmaxf(m, __shfl_xor(m, 16));
    float sum = 0.f;
    for (int wi = (lane >> 4); wi < ADJ_W; wi += 2) {
        unsigned w = adjs[rl * ADJ_W + wi];
        if (w == 0u) continue;
        const float* sb = s_lds + wi * 32;
        for (int b = 0; b < 32; ++b) {
            float s = lrelu(sb[b] + st);
            sum += ((w >> b) & 1u) ? __expf(s - m) : 0.f;
        }
    }
    sum += __shfl_xor(sum, 16);
    const float inv = 1.f / sum;

    v8f acc = {};
    const int koff = (lane >> 4) << 3;
    for (int jt = 0; jt < N_NODES / 32; ++jt) {
        const unsigned w = adjs[rl * ADJ_W + jt];
        if (__ballot(w != 0u) == 0ull) continue;
        const float4* sv = (const float4*)(s_lds + jt * 32 + koff);
        const float4 s0 = sv[0], s1 = sv[1];
        const float4 s2 = sv[4], s3 = sv[5];
        float sarr[16] = { s0.x, s0.y, s0.z, s0.w, s1.x, s1.y, s1.z, s1.w,
                           s2.x, s2.y, s2.z, s2.w, s3.x, s3.y, s3.z, s3.w };
        v16h a;
#pragma unroll
        for (int hh = 0; hh < 16; ++hh) {
            int K = hh + ((hh >> 3) << 3) + koff;
            float s = lrelu(sarr[hh] + st);
            float p = ((w >> K) & 1u) ? __expf(s - m) * inv : 0.f;
            a[hh] = (_Float16)p;
        }
        const v16h b = *(const v16h*)(f2p + (jt * 32 + lane) * 16);
        acc = __builtin_amdgcn_wmma_f32_16x16x32_f16(
            false, a, false, b, (short)0, acc, false, false);
    }
    const int col = lane & 15;
    if (col < NCLS) {
#pragma unroll
        for (int r = 0; r < 8; ++r) {
            int ro = rbase + r + ((lane >> 4) << 3);
            out[ro * NCLS + col] = acc[r];
        }
    }
}

// ---------------------------------------------------------------- host
extern "C" void kernel_launch(void* const* d_in, const int* in_sizes, int n_in,
                              void* d_out, int out_size, void* d_ws, size_t ws_size,
                              hipStream_t stream) {
    const float* x   = (const float*)d_in[0];
    const int*   ei  = (const int*)d_in[1];
    const float* W1  = (const float*)d_in[2];
    const float* b1  = (const float*)d_in[3];
    const float* aL1 = (const float*)d_in[4];
    const float* aR1 = (const float*)d_in[5];
    const float* W2  = (const float*)d_in[6];
    const float* b2  = (const float*)d_in[7];
    const float* aL2 = (const float*)d_in[8];
    const float* aR2 = (const float*)d_in[9];
    float* out = (float*)d_out;
    const int E = in_sizes[1] / 2;

    // workspace carve-up (256B-aligned offsets), total ~5.7 MB
    char* ws = (char*)d_ws;
    unsigned* adj   = (unsigned*)(ws + 0);          // 2,097,152
    _Float16* W1p   = (_Float16*)(ws + 2097152);    //    65,536
    float* feats1   = (float*)   (ws + 2162688);    // 1,048,576
    float* ssrc1T   = (float*)   (ws + 3211264);    //   131,072  [head][node]
    float* stgt1    = (float*)   (ws + 3342336);    //   131,072  [node][head]
    _Float16* f1p   = (_Float16*)(ws + 3473408);    // 1,048,576
    float* h1       = (float*)   (ws + 4521984);    // 1,048,576
    float* feats2   = (float*)   (ws + 5570560);    //   131,072 (114,688 used)
    float* ssrc2    = (float*)   (ws + 5701632);    //    16,384
    float* stgt2    = (float*)   (ws + 5718016);    //    16,384
    _Float16* f2p   = (_Float16*)(ws + 5734400);    //   131,072

    k_zero<<<2048, 256, 0, stream>>>(adj, N_NODES * ADJ_W);
    k_adj<<<(E + 255) / 256, 256, 0, stream>>>(adj, ei, ei + E, E);
    k_pack_w1<<<128, 256, 0, stream>>>(W1, W1p);
    k_gemm1<<<32, 256, 0, stream>>>(x, W1p, b1, feats1);
    k_scores1<<<128, 256, 0, stream>>>(feats1, aL1, aR1, ssrc1T, stgt1);
    k_pack_f1<<<2048, 256, 0, stream>>>(feats1, f1p);
    k_attn1<<<256, 256, 0, stream>>>(adj, ssrc1T, stgt1, f1p, h1);
    k_gemm2<<<16, 256, 0, stream>>>(h1, W2, b2, aL2, aR2, feats2, ssrc2, stgt2);
    k_pack_f2<<<256, 256, 0, stream>>>(feats2, f2p);
    k_attn2<<<256, 32, 0, stream>>>(adj, ssrc2, stgt2, f2p, out);
}
